// FlashAliBiCausalSelfAttention_9182640079355
// MI455X (gfx1250) — matrix operand order repaired and checked by reading, not compiled, not run on hardware
//
#include <hip/hip_runtime.h>
#include <math.h>

typedef _Float16 f16;
typedef __attribute__((ext_vector_type(16))) _Float16 v16h;
typedef __attribute__((ext_vector_type(8)))  _Float16 v8h;
typedef __attribute__((ext_vector_type(2)))  _Float16 v2h;
typedef __attribute__((ext_vector_type(8)))  float    v8f;

namespace {

constexpr int B_ = 2, L_ = 2048, D_ = 1024, H_ = 16, HD_ = 64;

__device__ __forceinline__ v8f v8f_zero() {
  v8f r;
  #pragma unroll
  for (int i = 0; i < 8; i++) r[i] = 0.0f;
  return r;
}

// A-fragment (16x32 f16) K-index for element pair base e=2*me (contiguous order):
// lanes 0-15: K 0..15; lanes 16-31: K 16..31 (matches load_b_frag).
__device__ __forceinline__ int a_koff(int me, int lhi) {
  int e = 2 * me;
  return e + (lhi << 4);
}

__device__ __forceinline__ v16h load_a_frag(const f16* rowbase, int lhi) {
  v16h a;
  #pragma unroll
  for (int me = 0; me < 8; me++) {
    v2h p = *(const v2h*)(rowbase + a_koff(me, lhi));
    a[2 * me] = p[0]; a[2 * me + 1] = p[1];
  }
  return a;
}

// base must already include n*stride + 16*lhi (B element e -> K = e + 16*lhi)
__device__ __forceinline__ v16h load_b_frag(const f16* base) {
  v16h b;
  #pragma unroll
  for (int me = 0; me < 8; me++) {
    v2h p = *(const v2h*)(base + 2 * me);
    b[2 * me] = p[0]; b[2 * me + 1] = p[1];
  }
  return b;
}

// 16B async global->LDS copy (ASYNCcnt-tracked; no VGPR data round trip).
// Flat LDS pointers: low 32 bits of the generic address are the LDS byte offset.
__device__ __forceinline__ void async_ld_b128(void* lds_dst, const void* gbase, int goff) {
  unsigned l = (unsigned)(uintptr_t)lds_dst;
  asm volatile("global_load_async_to_lds_b128 %0, %1, %2"
               :
               : "v"(l), "v"(goff), "s"(gbase)
               : "memory");
}

__device__ __forceinline__ void wait_async0() {
  asm volatile("s_wait_asynccnt 0x0" ::: "memory");
}

// -----------------------------------------------------------------------------
// WMMA GEMM: C[M,N] = A[M,K] * W[K,N] + bias[N]
// 256 threads = 8 waves; WG tile 64(M) x 256(N); wave tile 64x32 (8 WMMA/K-step);
// K-step 32; double-buffered LDS (one barrier per K-step).
// EPI==0: fp32 to outf [M,N].  EPI==1: scatter f16 q/k/v as [B,H,L,HD].
// (f32->f16 conversion during staging => async-to-LDS not applicable here.)
// -----------------------------------------------------------------------------
template <bool A_F16, int EPI>
__global__ __launch_bounds__(256) void gemm_wmma(
    const void* __restrict__ A_, const float* __restrict__ W,
    const float* __restrict__ bias, float* __restrict__ outf,
    f16* __restrict__ qb, f16* __restrict__ kb, f16* __restrict__ vb,
    int N, int K) {
  __shared__ __align__(16) f16 As[2][64 * 32];    // [m][k]
  __shared__ __align__(16) f16 Bs[2][256 * 32];   // [n][k] (transposed W tile)

  const int tid = threadIdx.x;
  const int lane = tid & 31;
  const int w = tid >> 5;
  const int lrow = lane & 15, lhi = lane >> 4;
  const int Mbase = blockIdx.y * 64;
  const int Nbase = blockIdx.x * 256;

  const int am = tid >> 2;         // 0..63
  const int ak = (tid & 3) * 8;    // 0..24
  const int bkk = tid >> 3;        // 0..31
  const int bn = (tid & 7) * 32;   // 0..224

  auto stage = [&](int p, int k0) {
    if (A_F16) {
      const f16* A = (const f16*)A_ + (size_t)(Mbase + am) * K + k0 + ak;
      *(v8h*)&As[p][am * 32 + ak] = *(const v8h*)A;
    } else {
      const float* A = (const float*)A_ + (size_t)(Mbase + am) * K + k0 + ak;
      float4 x0 = *(const float4*)A;
      float4 x1 = *(const float4*)(A + 4);
      f16* d = &As[p][am * 32 + ak];
      d[0] = (f16)x0.x; d[1] = (f16)x0.y; d[2] = (f16)x0.z; d[3] = (f16)x0.w;
      d[4] = (f16)x1.x; d[5] = (f16)x1.y; d[6] = (f16)x1.z; d[7] = (f16)x1.w;
    }
    const float* Wp = W + (size_t)(k0 + bkk) * N + Nbase + bn;
    #pragma unroll
    for (int i = 0; i < 32; i += 4) {
      float4 x = *(const float4*)(Wp + i);
      Bs[p][(bn + i + 0) * 32 + bkk] = (f16)x.x;
      Bs[p][(bn + i + 1) * 32 + bkk] = (f16)x.y;
      Bs[p][(bn + i + 2) * 32 + bkk] = (f16)x.z;
      Bs[p][(bn + i + 3) * 32 + bkk] = (f16)x.w;
    }
    // prefetch two K-steps ahead into L2
    if (k0 + 64 < K) {
      __builtin_prefetch(Wp + (size_t)64 * N, 0, 0);
      if (!A_F16)
        __builtin_prefetch((const float*)A_ + (size_t)(Mbase + am) * K + k0 + 64, 0, 0);
    }
  };

  v8f acc[4][2];
  #pragma unroll
  for (int i = 0; i < 4; i++)
    #pragma unroll
    for (int j = 0; j < 2; j++) acc[i][j] = v8f_zero();

  const int nks = K / 32;
  stage(0, 0);
  for (int kk = 0; kk < nks; kk++) {
    const int p = kk & 1;
    __syncthreads();                       // buf p staged; prior readers of buf p done
    if (kk + 1 < nks) stage(1 - p, (kk + 1) * 32);  // overlap with compute

    v16h a[4], b[2];
    #pragma unroll
    for (int i = 0; i < 4; i++)
      a[i] = load_a_frag(&As[p][(i * 16 + lrow) * 32], lhi);
    #pragma unroll
    for (int j = 0; j < 2; j++)
      b[j] = load_b_frag(&Bs[p][(w * 32 + j * 16 + lrow) * 32 + 16 * lhi]);
    #pragma unroll
    for (int i = 0; i < 4; i++)
      #pragma unroll
      for (int j = 0; j < 2; j++)
        acc[i][j] = __builtin_amdgcn_wmma_f32_16x16x32_f16(
            false, a[i], false, b[j], (short)0, acc[i][j], false, false);
  }

  // ---- epilogue ----
  #pragma unroll
  for (int i = 0; i < 4; i++)
    #pragma unroll
    for (int j = 0; j < 2; j++)
      #pragma unroll
      for (int r = 0; r < 8; r++) {
        const int row = Mbase + i * 16 + r + 8 * lhi;
        const int col = Nbase + w * 32 + j * 16 + lrow;
        const float v = acc[i][j][r] + bias[col];
        if (EPI == 0) {
          outf[(size_t)row * N + col] = v;
        } else {
          const int which = col >> 10;         // 0:q 1:k 2:v
          const int h = (col >> 6) & 15;
          const int hd = col & 63;
          const int bidx = row >> 11;          // / L
          const int l = row & (L_ - 1);
          f16* dst = (which == 0) ? qb : (which == 1) ? kb : vb;
          dst[(((size_t)bidx * H_ + h) * L_ + l) * HD_ + hd] = (f16)v;
        }
      }
}

// -----------------------------------------------------------------------------
// Flash attention with ALiBi + causal mask.
// Grid: (L/128, B*H). 8 waves; wave owns 16 query rows.
// Q + K tiles staged via GLOBAL_LOAD_ASYNC_TO_LDS_B128 (ASYNCcnt-fenced),
// V transposed via VGPR path; double-buffered; one barrier per key tile.
// -----------------------------------------------------------------------------
__global__ __launch_bounds__(256) void attn_wmma(
    const f16* __restrict__ qb, const f16* __restrict__ kb,
    const f16* __restrict__ vb, f16* __restrict__ ob) {
  __shared__ __align__(16) f16 Ks[2][32 * 64];    // [key][dim]
  __shared__ __align__(16) f16 Vt[2][64 * 32];    // [dim][key]
  __shared__ __align__(16) f16 Ps[8][16 * 32];    // per-wave P tile [q][key]
  __shared__ __align__(16) f16 Qs[8 * 16 * 64];   // 128 q rows, contiguous

  const int tid = threadIdx.x;
  const int lane = tid & 31;
  const int w = tid >> 5;
  const int lrow = lane & 15, lhi = lane >> 4;
  const int bh = blockIdx.y;
  const int b = bh >> 4, h = bh & 15;
  const int qwg = blockIdx.x * 128;
  // wave-uniform scalar: keeps the causal skip a scalar branch (EXEC untouched)
  const int q0 = __builtin_amdgcn_readfirstlane(qwg + w * 16);

  const size_t headoff = (size_t)(b * H_ + h) * L_ * HD_;
  const f16* kbasep = kb + headoff;
  const f16* vbasep = vb + headoff;

  auto stageKV = [&](int p, int kbase) {
    // K tile: 4KB contiguous copy, one async b128 per thread
    async_ld_b128((char*)&Ks[p][0] + tid * 16, kbasep + (size_t)kbase * HD_, tid * 16);
    // V tile: transpose through VGPRs (v2h loads -> two b16 LDS stores)
    const f16* srcv = vbasep + (size_t)kbase * HD_;
    #pragma unroll
    for (int i = 0; i < 4; i++) {
      const int idx2 = (i * 256 + tid) * 2;
      v2h x = *(const v2h*)(srcv + idx2);
      const int key = idx2 >> 6, d = idx2 & 63;
      Vt[p][d * 32 + key] = x[0];
      Vt[p][(d + 1) * 32 + key] = x[1];
    }
    // prefetch the tile after this one into L2 (one 128B line per lane)
    if (kbase + 64 <= L_ && tid < 32) {
      __builtin_prefetch(kbasep + (size_t)(kbase + 32) * HD_ + tid * 64, 0, 0);
      __builtin_prefetch(srcv + 32 * HD_ + tid * 64, 0, 0);
    }
  };

  // ---- preamble: async-stage Q (16KB contiguous) + first KV tile ----
  #pragma unroll
  for (int i = 0; i < 4; i++)
    async_ld_b128((char*)Qs + tid * 64 + i * 16,
                  qb + headoff + (size_t)qwg * HD_, tid * 64 + i * 16);
  stageKV(0, 0);
  wait_async0();
  __syncthreads();

  v16h qf[2];
  #pragma unroll
  for (int c = 0; c < 2; c++)
    qf[c] = load_a_frag(&Qs[(w * 16 + lrow) * 64 + c * 32], lhi);

  float m_run[8], l_part[8];
  v8f o[4];
  #pragma unroll
  for (int r = 0; r < 8; r++) { m_run[r] = -INFINITY; l_part[r] = 0.0f; }
  #pragma unroll
  for (int c = 0; c < 4; c++) o[c] = v8f_zero();

  const float slope = exp2f(-0.5f * (float)(h + 1));  // ALiBi, H=16
  const float invs = 0.125f;                          // 1/sqrt(64)

  const int nkt = (qwg >> 5) + 4;  // key tiles covering 0..qwg+127
  for (int kt = 0; kt < nkt; kt++) {
    const int kbase = kt * 32;
    const int p = kt & 1;
    if (kt + 1 < nkt) stageKV(1 - p, (kt + 1) * 32);  // overlap with compute

    if (kbase <= q0 + 15) {  // scalar, wave-uniform: EXEC stays all-ones for WMMA
      // ---- S = Q * K^T (two 16-key column tiles, HD=64 split over 2 WMMAs) ----
      v8f s[2];
      s[0] = v8f_zero(); s[1] = v8f_zero();
      #pragma unroll
      for (int j = 0; j < 2; j++)
        #pragma unroll
        for (int c = 0; c < 2; c++) {
          v16h bk_ = load_b_frag(&Ks[p][(j * 16 + lrow) * 64 + c * 32 + 16 * lhi]);
          s[j] = __builtin_amdgcn_wmma_f32_16x16x32_f16(
              false, qf[c], false, bk_, (short)0, s[j], false, false);
        }

      // ---- scale + ALiBi (incremental) + causal mask + online softmax ----
      // dq0 = key0 - qrow at r=0; per row r: key0-qrow = dq0 - r.
      const int dq0 = kbase + lrow - q0 - 8 * lhi;
      float al = slope * (float)dq0;          // ALiBi bias, updated by -slope per row
      const float al16 = 16.0f * slope;       // sub-tile 1 is 16 keys further
      #pragma unroll
      for (int r = 0; r < 8; r++) {
        float v0 = s[0][r] * invs + al;
        float v1 = s[1][r] * invs + al + al16;
        if (dq0 > r) v0 = -INFINITY;          // key0 > qrow
        if (dq0 + 16 > r) v1 = -INFINITY;     // key1 > qrow
        al -= slope;
        float mt = fmaxf(v0, v1);
        #pragma unroll
        for (int off = 1; off < 16; off <<= 1)
          mt = fmaxf(mt, __shfl_xor(mt, off, 32));   // row-uniform running max
        const float nm = fmaxf(m_run[r], mt);
        const float alpha = __expf(m_run[r] - nm);   // row-uniform
        m_run[r] = nm;
        const float p0 = __expf(v0 - nm);
        const float p1 = __expf(v1 - nm);
        // lane-partial row sum: alpha is row-uniform, so partials stay consistent
        l_part[r] = l_part[r] * alpha + (p0 + p1);
        #pragma unroll
        for (int c = 0; c < 4; c++) o[c][r] *= alpha;
        Ps[w][(r + 8 * lhi) * 32 + lrow] = (f16)p0;
        Ps[w][(r + 8 * lhi) * 32 + 16 + lrow] = (f16)p1;
      }

      // ---- P (D-layout) -> A-fragment via wave-local LDS round trip ----
      v16h pf = load_a_frag(&Ps[w][lrow * 32], lhi);

      // ---- O += P * V ----
      #pragma unroll
      for (int c = 0; c < 4; c++) {
        v16h bv = load_b_frag(&Vt[p][(c * 16 + lrow) * 32 + 16 * lhi]);
        o[c] = __builtin_amdgcn_wmma_f32_16x16x32_f16(
            false, pf, false, bv, (short)0, o[c], false, false);
      }
    }

    if (kt + 1 < nkt) {      // publish next buffer (uniform condition)
      wait_async0();         // this wave's async K writes complete
      __syncthreads();       // all waves' staging visible; readers of old buf done
    }
  }

  // ---- single end reduction of row sums, normalize, write [B, L, H*HD] ----
  float invl[8];
  #pragma unroll
  for (int r = 0; r < 8; r++) {
    float rs = l_part[r];
    #pragma unroll
    for (int off = 1; off < 16; off <<= 1)
      rs += __shfl_xor(rs, off, 32);
    invl[r] = 1.0f / rs;
  }
  #pragma unroll
  for (int c = 0; c < 4; c++)
    #pragma unroll
    for (int r = 0; r < 8; r++) {
      const int q = q0 + r + 8 * lhi;
      const int dim = c * 16 + lrow;
      ob[((size_t)b * L_ + q) * D_ + h * HD_ + dim] = (f16)(o[c][r] * invl[r]);
    }
}

}  // namespace

extern "C" void kernel_launch(void* const* d_in, const int* in_sizes, int n_in,
                              void* d_out, int out_size, void* d_ws, size_t ws_size,
                              hipStream_t stream) {
  const float* x     = (const float*)d_in[0];
  const float* W_in  = (const float*)d_in[1];
  const float* b_in  = (const float*)d_in[2];
  const float* W_out = (const float*)d_in[3];
  const float* b_out = (const float*)d_in[4];
  float* out = (float*)d_out;

  // Workspace: q/k/v f16 [B,H,L,HD] = 8 MiB each; attn out f16 [B,L,D] = 8 MiB.
  char* ws = (char*)d_ws;
  f16* qb = (f16*)(ws);
  f16* kb = (f16*)(ws + (size_t)(8u << 20));
  f16* vb = (f16*)(ws + (size_t)(16u << 20));
  f16* ob = (f16*)(ws + (size_t)(24u << 20));

  const int M = B_ * L_;  // 4096

  // 1) QKV projection: [4096,1024] x [1024,3072] -> f16 q/k/v
  gemm_wmma<false, 1><<<dim3((3 * D_) / 256, M / 64), 256, 0, stream>>>(
      x, W_in, b_in, nullptr, qb, kb, vb, 3 * D_, D_);

  // 2) Flash ALiBi causal attention
  attn_wmma<<<dim3(L_ / 128, B_ * H_), 256, 0, stream>>>(qb, kb, vb, ob);

  // 3) Output projection: [4096,1024] x [1024,1024] -> fp32 d_out
  gemm_wmma<true, 0><<<dim3(D_ / 256, M / 64), 256, 0, stream>>>(
      ob, W_out, b_out, out, nullptr, nullptr, nullptr, D_, D_);
}
